// StaticSpexphormerAttention_22436909154859
// MI455X (gfx1250) — matrix operand order repaired and checked by reading, not compiled, not run on hardware
//
#include <hip/hip_runtime.h>
#include <hip/hip_bf16.h>
#include <cstdint>

// ---------------------------------------------------------------------------
// StaticSpexphormerAttention for MI455X (gfx1250, wave32, WMMA 16x16x32 bf16)
//
// N_NODES=10000, DEG=32, IN_DIM=OUT_DIM=256, N_HEADS=8, HEAD_DIM=32
//
//  k0: pack weights f32 -> bf16 in WMMA B-fragment order (L2-resident reuse)
//  k1: QKV = x @ [WQ|WK|WV]   (WMMA, 16-node M-tiles, 48 N-tiles)
//  k2: per node: E1/E2 GEMM (WMMA from LDS-staged bf16 edge rows) fused with
//      gather + score + softmax(deg=32 == wave32) + attn@V epilogue
// ---------------------------------------------------------------------------

#define N_NODES 10000
#define DEG     32
#define DIM     256
#define N_HEADS 8
#define HEAD_DIM 32

typedef __attribute__((ext_vector_type(16))) __bf16 v16bf;
typedef __attribute__((ext_vector_type(8)))  float  v8f;
typedef __attribute__((ext_vector_type(8)))  unsigned int v8u;

__device__ __forceinline__ uint32_t bf16pair(float a, float b) {
  uint32_t ua = __builtin_bit_cast(uint32_t, a);
  uint32_t ub = __builtin_bit_cast(uint32_t, b);
  ua = (ua + 0x7FFFu + ((ua >> 16) & 1u)) >> 16;   // RNE
  ub = (ub + 0x7FFFu + ((ub >> 16) & 1u)) >> 16;
  return (ub << 16) | (ua & 0xFFFFu);
}

__device__ __forceinline__ v8f wmma_bf16(v8u a, v8u b, v8f c) {
  return __builtin_amdgcn_wmma_f32_16x16x32_bf16(
      false, __builtin_bit_cast(v16bf, a),
      false, __builtin_bit_cast(v16bf, b),
      (short)0, c, false, false);
}

// A fragment (16x32 bf16, M x K) from row-major bf16 tile in LDS.
// Layout: lanes 0-15 M=0..15 K-base 0, lanes 16-31 same M, K-base +16;
// VGPR i in 0..7 holds K = kb+2i,kb+2i+1 (matches packed B).
__device__ __forceinline__ v8u load_a_frag(const uint32_t* lds, int m, int kt,
                                           int rowStrideDw) {
  int lane = threadIdx.x & 31;
  const uint32_t* p = lds + m * rowStrideDw + kt * 16 + ((lane & 16) ? 8 : 0);
  v8u a;
  a[0] = p[0]; a[1] = p[1]; a[2] = p[2]; a[3] = p[3];
  a[4] = p[4]; a[5] = p[5]; a[6] = p[6]; a[7] = p[7];
  return a;
}

// B fragment (32x16 bf16, K x N) pre-swizzled in global: 8 contiguous dwords
// per lane -> two global_load_b128.
__device__ __forceinline__ v8u load_b_frag(const uint32_t* wp, int nt, int kt) {
  int lane = threadIdx.x & 31;
  const uint32_t* p = wp + (((size_t)(nt * 8 + kt) * 32 + lane) << 3);
  v8u b;
#pragma unroll
  for (int j = 0; j < 8; ++j) b[j] = p[j];
  return b;
}

// --------------------------- kernel 0: weight pack -------------------------
// out[((nt*8+kt)*32+L)*8+j] = pack(W[kt*32+kb(L)+2j][nt*16+(L&15)], ...+1)
// kb(L) = (L&16)?16:0  (contiguous k order, same as A: lanes 16-31 hold K=16..31)
__global__ void pack_weight_kernel(const float* __restrict__ W, int N,
                                   int Ntiles, uint32_t* __restrict__ out) {
  int t = blockIdx.x * blockDim.x + threadIdx.x;
  int total = Ntiles * 2048;  // Ntiles * 8 ktiles * 32 lanes * 8 dwords
  if (t >= total) return;
  int j  = t & 7;
  int L  = (t >> 3) & 31;
  int kt = (t >> 8) & 7;
  int nt = t >> 11;
  int n  = nt * 16 + (L & 15);
  int k0 = kt * 32 + ((L & 16) ? 16 : 0) + 2 * j;
  float f0 = (n < N) ? W[(size_t)k0 * N + n]       : 0.0f;
  float f1 = (n < N) ? W[(size_t)(k0 + 1) * N + n] : 0.0f;
  out[t] = bf16pair(f0, f1);
}

// --------------------------- kernel 1: QKV GEMM ----------------------------
// grid = 625 blocks (16 nodes each), 256 threads = 8 waves.
// Wave w sweeps N-tiles {w, w+8, ...} over concatenated [Q|K|V] (48 tiles).
__global__ void __launch_bounds__(256)
qkv_gemm_kernel(const float* __restrict__ x,
                const uint32_t* __restrict__ wpackQ,  // Q,K,V packs contiguous
                float* __restrict__ qkv) {            // [3][N_NODES][256]
  __shared__ uint32_t xs[16 * 128];  // 16 rows x 256 bf16 (8 KB)

  int t = threadIdx.x;
  int row = t >> 4;
  int c0 = (t & 15) * 16;
  const float* xr = x + ((size_t)(blockIdx.x * 16 + row)) * DIM + c0;
  uint32_t* dst = xs + row * 128 + (c0 >> 1);
#pragma unroll
  for (int i = 0; i < 16; i += 2) dst[i >> 1] = bf16pair(xr[i], xr[i + 1]);
  __syncthreads();

  int w = t >> 5, lane = t & 31;
  int m = lane & 15;
  for (int ntg = w; ntg < 48; ntg += 8) {
    int sel = ntg >> 4;        // 0=Q 1=K 2=V
    int nt  = ntg & 15;
    const uint32_t* wp = wpackQ + (size_t)sel * 32768;
    v8f acc = {};
#pragma unroll
    for (int kt = 0; kt < 8; ++kt) {
      v8u a = load_a_frag(xs, m, kt, 128);
      v8u b = load_b_frag(wp, nt, kt);
      acc = wmma_bf16(a, b, acc);
    }
    // D layout: VGPR r -> (M=r, N=lane) lanes 0-15 ; (M=8+r, N=lane-16)
    int cn = lane & 15;
    int rb = (lane & 16) ? 8 : 0;
    float* op = qkv + (size_t)sel * (N_NODES * DIM) +
                ((size_t)blockIdx.x * 16 + rb) * DIM + nt * 16 + cn;
#pragma unroll
    for (int r = 0; r < 8; ++r) op[(size_t)r * DIM] = acc[r];
  }
}

// ------------------- kernel 2: fused edge GEMM + attention -----------------
// grid = N_NODES blocks, 256 threads = 8 waves (one wave per head later).
__global__ void __launch_bounds__(256)
fused_edge_attn_kernel(const float* __restrict__ edge_attr,
                       const long long* __restrict__ edge_src,  // edge_index[0]
                       const uint32_t* __restrict__ we1pack,
                       const uint32_t* __restrict__ we2pack,
                       const float* __restrict__ bE2,
                       const float* __restrict__ qkv,  // [3][N_NODES][256]
                       float* __restrict__ out) {
  __shared__ uint32_t es[32 * 128];   // 32 edge rows x 256 bf16 (16 KB)
  __shared__ float e1s[32 * 256];     // E1 tile (32 KB)
  __shared__ float e2s[32 * 8];       // E2 + bias
  __shared__ float qs[256];           // Q row of this node
  __shared__ float attns[8 * 32];     // softmax weights per head
  __shared__ int   nbrs[32];

  const int n = blockIdx.x;
  const int t = threadIdx.x;
  const size_t e0 = (size_t)n * DEG;

  const float* Qw = qkv;
  const float* Kw = qkv + (size_t)1 * N_NODES * DIM;
  const float* Vw = qkv + (size_t)2 * N_NODES * DIM;

  if (t < 32) nbrs[t] = (int)edge_src[e0 + t];
  qs[t] = Qw[(size_t)n * DIM + t];

  { // stage 32 edge rows -> bf16 LDS
    int row = t >> 3;
    int c0 = (t & 7) * 32;
    const float* ep = edge_attr + (e0 + row) * DIM + c0;
    uint32_t* dst = es + row * 128 + (c0 >> 1);
#pragma unroll
    for (int i = 0; i < 32; i += 2) dst[i >> 1] = bf16pair(ep[i], ep[i + 1]);
  }
  __syncthreads();

  // 34 output tiles: 2 M-tiles x (16 E1 N-tiles + 1 padded E2 N-tile)
  int w = t >> 5, lane = t & 31;
  for (int tile = w; tile < 34; tile += 8) {
    int mt  = tile & 1;
    int ntg = tile >> 1;                  // 0..16
    const uint32_t* wp = (ntg < 16) ? we1pack : we2pack;
    int nt = (ntg < 16) ? ntg : 0;
    int m = mt * 16 + (lane & 15);
    v8f acc = {};
#pragma unroll
    for (int kt = 0; kt < 8; ++kt) {
      v8u a = load_a_frag(es, m, kt, 128);
      v8u b = load_b_frag(wp, nt, kt);
      acc = wmma_bf16(a, b, acc);
    }
    int cn = lane & 15;
    int rb = mt * 16 + ((lane & 16) ? 8 : 0);
    if (ntg < 16) {
#pragma unroll
      for (int r = 0; r < 8; ++r) e1s[(rb + r) * 256 + ntg * 16 + cn] = acc[r];
    } else if (cn < N_HEADS) {
      float bias = bE2[cn];
#pragma unroll
      for (int r = 0; r < 8; ++r) e2s[(rb + r) * 8 + cn] = acc[r] + bias;
    }
  }
  __syncthreads();

  // ---- attention: wave = head h, lane = edge d (DEG == wave32, exact fit)
  int h = w, d = lane;
  {
    const float* Krow = Kw + (size_t)nbrs[d] * DIM + h * HEAD_DIM;
    const float* qh = qs + h * HEAD_DIM;
    const float* e1r = e1s + d * 256 + h * HEAD_DIM;
    float s = 0.0f;
#pragma unroll
    for (int k = 0; k < HEAD_DIM; ++k) s += e1r[k] * Krow[k] * qh[k];
    s += e2s[d * 8 + h];
    s = fminf(8.0f, fmaxf(-8.0f, s));
    // wave-wide softmax over the 32 edges
    float mx = s;
#pragma unroll
    for (int off = 16; off > 0; off >>= 1)
      mx = fmaxf(mx, __shfl_xor(mx, off, 32));
    float e = __expf(s - mx);
    float sum = e;
#pragma unroll
    for (int off = 16; off > 0; off >>= 1) sum += __shfl_xor(sum, off, 32);
    attns[h * 32 + d] = e / sum;
  }
  __syncthreads();

  // ---- output: wave = head h, lane = channel k ; coalesced V gathers
  {
    int k = lane;
    float o = 0.0f;
#pragma unroll 8
    for (int d2 = 0; d2 < DEG; ++d2)
      o += attns[h * 32 + d2] * Vw[(size_t)nbrs[d2] * DIM + h * HEAD_DIM + k];
    out[(size_t)n * DIM + h * HEAD_DIM + k] = o;
  }
}

// ---------------------------------------------------------------------------
extern "C" void kernel_launch(void* const* d_in, const int* in_sizes, int n_in,
                              void* d_out, int out_size, void* d_ws,
                              size_t ws_size, hipStream_t stream) {
  const float*     x         = (const float*)d_in[0];
  const long long* edge_idx  = (const long long*)d_in[1];  // int64 [2][E]
  const float*     edge_attr = (const float*)d_in[2];
  const float*     WQ        = (const float*)d_in[3];
  const float*     WK        = (const float*)d_in[4];
  const float*     WV        = (const float*)d_in[5];
  const float*     WE1       = (const float*)d_in[6];
  const float*     WE2       = (const float*)d_in[7];
  const float*     bE2       = (const float*)d_in[8];
  float*           out       = (float*)d_out;

  // workspace layout (dwords)
  uint32_t* wqP  = (uint32_t*)d_ws;      // 32768 dw each for Q/K/V/E1
  uint32_t* wkP  = wqP + 32768;
  uint32_t* wvP  = wkP + 32768;
  uint32_t* we1P = wvP + 32768;
  uint32_t* we2P = we1P + 32768;         // 2048 dw (1 padded N-tile)
  float*    qkv  = (float*)(we2P + 2048);  // [3][N_NODES][256] f32

  // k0: pack all five weights (f32 -> bf16 fragment order)
  pack_weight_kernel<<<(16 * 2048 + 255) / 256, 256, 0, stream>>>(WQ, DIM, 16, wqP);
  pack_weight_kernel<<<(16 * 2048 + 255) / 256, 256, 0, stream>>>(WK, DIM, 16, wkP);
  pack_weight_kernel<<<(16 * 2048 + 255) / 256, 256, 0, stream>>>(WV, DIM, 16, wvP);
  pack_weight_kernel<<<(16 * 2048 + 255) / 256, 256, 0, stream>>>(WE1, DIM, 16, we1P);
  pack_weight_kernel<<<(1 * 2048 + 255) / 256, 256, 0, stream>>>(WE2, N_HEADS, 1, we2P);

  // k1: QKV projections (625 M-tiles of 16 nodes)
  qkv_gemm_kernel<<<N_NODES / 16, 256, 0, stream>>>(x, wqP, qkv);

  // k2: fused edge GEMM + attention (one node per block)
  fused_edge_attn_kernel<<<N_NODES, 256, 0, stream>>>(
      edge_attr, edge_idx /* row 0 = sources */, we1P, we2P, bE2, qkv, out);
}
